// CausalSelfAttention_523986010332
// MI455X (gfx1250) — matrix operand order repaired and checked by reading, not compiled, not run on hardware
//
#include <hip/hip_runtime.h>

// ---------------------------------------------------------------------------
// CDNA5 (gfx1250) causal self-attention: bf16 WMMA everywhere, f32 accumulate.
// B=2, T=2048, C=1024, H=16, HD=64.
// ---------------------------------------------------------------------------

typedef __bf16 bf16;
typedef __bf16 bf16x4 __attribute__((ext_vector_type(4)));
typedef __bf16 bf16x8 __attribute__((ext_vector_type(8)));
typedef __bf16 v16bf  __attribute__((ext_vector_type(16)));
typedef float  v8f    __attribute__((ext_vector_type(8)));
typedef int    v4i    __attribute__((ext_vector_type(4)));

#define AS1 __attribute__((address_space(1)))
#define AS3 __attribute__((address_space(3)))

#if defined(__has_builtin)
#if __has_builtin(__builtin_amdgcn_global_load_async_to_lds_b128)
#define HAVE_ASYNC_LDS 1
#endif
#endif

static __device__ __forceinline__ void async_copy_b128(const bf16* gsrc, bf16* ldst) {
#if defined(HAVE_ASYNC_LDS)
  // (global v4i*, lds v4i*, imm offset, imm cpol) -> global_load_async_to_lds_b128
  __builtin_amdgcn_global_load_async_to_lds_b128((AS1 v4i*)gsrc, (AS3 v4i*)ldst, 0, 0);
#else
  *(bf16x8*)ldst = *(const bf16x8*)gsrc;
#endif
}

static __device__ __forceinline__ void async_wait0() {
#if defined(HAVE_ASYNC_LDS)
#if __has_builtin(__builtin_amdgcn_s_wait_asynccnt)
  __builtin_amdgcn_s_wait_asynccnt(0);
#else
  asm volatile("s_wait_asynccnt 0x0" ::: "memory");
#endif
#endif
}

union Frag16 { v16bf v; bf16x8 h[2]; };

static __device__ __forceinline__ v8f wmma_bf16(const Frag16& a, const Frag16& b, v8f c) {
  // D = A(16x32 bf16) * B(32x16 bf16) + C(16x16 f32)
  return __builtin_amdgcn_wmma_f32_16x16x32_bf16(false, a.v, false, b.v,
                                                 (short)0, c, false, false);
}

// ---------------------------------------------------------------------------
// Kernel 1: qkv = x @ w_qkv + b_qkv, scattered into Q [BH,T,64], K [BH,T,64],
// V^T [BH,64,T] as bf16.  M=4096, N=3072, K=1024.  Block tile 128x128x32.
// ---------------------------------------------------------------------------
__global__ __launch_bounds__(256) void qkv_proj_kernel(
    const float* __restrict__ x, const float* __restrict__ w,
    const float* __restrict__ bias,
    bf16* __restrict__ qB, bf16* __restrict__ kB, bf16* __restrict__ vtB) {
  __shared__ bf16 As[128 * 40];   // [m][k], stride 40 (pad -> 16B aligned rows)
  __shared__ bf16 Bs[128 * 40];   // [n][k] transposed

  const int tid  = threadIdx.x;
  const int lane = tid & 31, wid = tid >> 5;
  const int l16  = lane & 15, lhi = lane >> 4;
  const int waveM = wid & 3, waveN = wid >> 2;       // 4x2 wave grid
  const int mBase = (blockIdx.x & 31) * 128;         // 32 M blocks
  const int nBase = (blockIdx.x >> 5) * 128;         // 24 N blocks

  v8f acc[2][4] = {};

  for (int k0 = 0; k0 < 1024; k0 += 32) {
    __syncthreads();
    // Prefetch next K-tile while this one is being staged/consumed.
    if (k0 + 32 < 1024) {
      __builtin_prefetch(x + (size_t)(mBase + (tid >> 3)) * 1024 + (k0 + 32) + ((tid & 7) << 2), 0, 1);
      __builtin_prefetch(w + (size_t)(k0 + 32 + (tid >> 5)) * 3072 + nBase + ((tid & 31) << 2), 0, 1);
    }
    // Stage A tile: 128x32 f32 -> bf16.  1024 float4 loads, 4 per thread.
    #pragma unroll
    for (int i = 0; i < 4; ++i) {
      int li = tid + i * 256;
      int m = li >> 3, kq = (li & 7) << 2;
      float4 f = *(const float4*)(x + (size_t)(mBase + m) * 1024 + k0 + kq);
      bf16x4 hv; hv[0]=(bf16)f.x; hv[1]=(bf16)f.y; hv[2]=(bf16)f.z; hv[3]=(bf16)f.w;
      *(bf16x4*)(As + m * 40 + kq) = hv;
    }
    // Stage B tile: 32x128 f32 (row-major [K][N]) -> transposed bf16 [n][k].
    #pragma unroll
    for (int i = 0; i < 4; ++i) {
      int li = tid + i * 256;
      int kk = li >> 5, nq = (li & 31) << 2;
      float4 f = *(const float4*)(w + (size_t)(k0 + kk) * 3072 + nBase + nq);
      Bs[(nq + 0) * 40 + kk] = (bf16)f.x;
      Bs[(nq + 1) * 40 + kk] = (bf16)f.y;
      Bs[(nq + 2) * 40 + kk] = (bf16)f.z;
      Bs[(nq + 3) * 40 + kk] = (bf16)f.w;
    }
    __syncthreads();

    Frag16 a[2], b[4];
    #pragma unroll
    for (int mr = 0; mr < 2; ++mr) {                  // A frag: lanes>=16 -> K+8
      int row = waveM * 32 + mr * 16 + l16;
      int d0 = lhi ? 8 : 0;
      a[mr].h[0] = *(const bf16x8*)(As + row * 40 + d0);
      a[mr].h[1] = *(const bf16x8*)(As + row * 40 + d0 + 16);
    }
    #pragma unroll
    for (int nr = 0; nr < 4; ++nr) {                  // B frag: lanes>=16 -> K+8
      int col = waveN * 64 + nr * 16 + l16;
      int d0 = lhi ? 8 : 0;
      b[nr].h[0] = *(const bf16x8*)(Bs + col * 40 + d0);
      b[nr].h[1] = *(const bf16x8*)(Bs + col * 40 + d0 + 16);
    }
    #pragma unroll
    for (int mr = 0; mr < 2; ++mr)
      #pragma unroll
      for (int nr = 0; nr < 4; ++nr)
        acc[mr][nr] = wmma_bf16(a[mr], b[nr], acc[mr][nr]);
  }

  // Epilogue: +bias, scatter into Q/K/V^T as bf16.
  #pragma unroll
  for (int nr = 0; nr < 4; ++nr) {
    int n = nBase + waveN * 64 + nr * 16 + l16;
    float bv = bias[n];
    int h = n / 192, r = n % 192;                     // head, slot within head
    #pragma unroll
    for (int mr = 0; mr < 2; ++mr) {
      #pragma unroll
      for (int vv = 0; vv < 8; ++vv) {
        int m = mBase + waveM * 32 + mr * 16 + vv + (lhi << 3);
        int bidx = m >> 11, t = m & 2047;
        size_t bh = (size_t)bidx * 16 + h;
        bf16 hv = (bf16)(acc[mr][nr][vv] + bv);
        if (r < 64)       qB[(bh * 2048 + t) * 64 + r] = hv;
        else if (r < 128) kB[(bh * 2048 + t) * 64 + (r - 64)] = hv;
        else              vtB[(bh * 64 + (r - 128)) * 2048 + t] = hv;
      }
    }
  }
}

// ---------------------------------------------------------------------------
// Kernel 2: flash attention.  Block = (qb, bh); 8 waves x 16 query rows.
// KV tiles of 64 staged in LDS via ASYNC loads; online softmax in C-layout.
// ---------------------------------------------------------------------------
__global__ __launch_bounds__(256) void flash_attn_kernel(
    const bf16* __restrict__ qBuf, const bf16* __restrict__ kBuf,
    const bf16* __restrict__ vtBuf, bf16* __restrict__ oBuf) {
  __shared__ bf16 ks[64 * 72];          // [j][d]  (K tile, natural layout)
  __shared__ bf16 vs[64 * 72];          // [d][j]  (V^T tile)
  __shared__ bf16 ps[8 * 16 * 72];      // per-wave P scratch [16][64]

  const int tid  = threadIdx.x, lane = tid & 31, wid = tid >> 5;
  const int l16  = lane & 15, lhi = lane >> 4;
  const int qb0  = blockIdx.x * 128;
  const size_t bh = blockIdx.y;
  const bf16* qg = qBuf  + bh * (size_t)(2048 * 64);
  const bf16* kg = kBuf  + bh * (size_t)(2048 * 64);
  const bf16* vg = vtBuf + bh * (size_t)(64 * 2048);
  bf16*       og = oBuf  + bh * (size_t)(2048 * 64);

  // Q fragments: two 16x32 A-frags covering hd=64 (kept in VGPRs for all KV).
  Frag16 qa[2];
  {
    int row = qb0 + wid * 16 + l16;
    int d0 = lhi ? 8 : 0;
    #pragma unroll
    for (int ksp = 0; ksp < 2; ++ksp) {
      qa[ksp].h[0] = *(const bf16x8*)(qg + (size_t)row * 64 + ksp * 32 + d0);
      qa[ksp].h[1] = *(const bf16x8*)(qg + (size_t)row * 64 + ksp * 32 + d0 + 16);
    }
  }

  float mrow[8], lsum[8];
  v8f accO[4] = {};
  #pragma unroll
  for (int i = 0; i < 8; ++i) { mrow[i] = -1e30f; lsum[i] = 0.0f; }

  const int jEnd  = qb0 + 128;          // causal: only tiles up to the diagonal
  const int pbase = wid * (16 * 72);

  for (int jt = 0; jt < jEnd; jt += 64) {
    __syncthreads();
    // Stage K (64x64) and V^T (64x64) tiles: pure bf16 copies -> async to LDS.
    #pragma unroll
    for (int i = 0; i < 2; ++i) {
      int li = tid + i * 256;
      int j = li >> 3, d = (li & 7) << 3;
      async_copy_b128(kg + (size_t)(jt + j) * 64 + d, ks + j * 72 + d);
    }
    #pragma unroll
    for (int i = 0; i < 2; ++i) {
      int li = tid + i * 256;
      int d = li >> 3, jq = (li & 7) << 3;
      async_copy_b128(vg + (size_t)d * 2048 + jt + jq, vs + d * 72 + jq);
    }
    async_wait0();                      // our wave's async copies landed in LDS
    __syncthreads();                    // -> everyone's copies visible

    // S = Q * K^T   (16 x 64 tile of scores, 8 WMMAs)
    v8f accS[4] = {};
    #pragma unroll
    for (int ksp = 0; ksp < 2; ++ksp) {
      #pragma unroll
      for (int jn = 0; jn < 4; ++jn) {
        Frag16 kf;
        int col = jn * 16 + l16;
        int dd = ksp * 32 + (lhi ? 8 : 0);
        kf.h[0] = *(const bf16x8*)(ks + col * 72 + dd);
        kf.h[1] = *(const bf16x8*)(ks + col * 72 + dd + 16);
        accS[jn] = wmma_bf16(qa[ksp], kf, accS[jn]);
      }
    }

    // Online softmax: rows live across the 16 lanes of each half-wave.
    #pragma unroll
    for (int vv = 0; vv < 8; ++vv) {
      int rg = qb0 + wid * 16 + vv + (lhi << 3);      // global query row
      float sv[4];
      #pragma unroll
      for (int jn = 0; jn < 4; ++jn) {
        int col = jt + jn * 16 + l16;
        float s = accS[jn][vv] * 0.125f;              // 1/sqrt(64)
        sv[jn] = (col > rg) ? -1e30f : s;             // causal mask
      }
      float mx = fmaxf(fmaxf(sv[0], sv[1]), fmaxf(sv[2], sv[3]));
      #pragma unroll
      for (int off = 8; off >= 1; off >>= 1)
        mx = fmaxf(mx, __shfl_xor(mx, off, 16));
      float mnew  = fmaxf(mrow[vv], mx);
      float alpha = __expf(mrow[vv] - mnew);
      mrow[vv] = mnew;
      float p[4], psum = 0.0f;
      #pragma unroll
      for (int jn = 0; jn < 4; ++jn) { p[jn] = __expf(sv[jn] - mnew); psum += p[jn]; }
      #pragma unroll
      for (int off = 8; off >= 1; off >>= 1)
        psum += __shfl_xor(psum, off, 16);
      lsum[vv] = lsum[vv] * alpha + psum;
      #pragma unroll
      for (int dn = 0; dn < 4; ++dn) accO[dn][vv] *= alpha;
      // C-layout -> LDS [row][col] so it can be reloaded in A-layout.
      int rl = vv + (lhi << 3);
      #pragma unroll
      for (int jn = 0; jn < 4; ++jn)
        ps[pbase + rl * 72 + jn * 16 + l16] = (bf16)p[jn];
    }

    // O += P * V   (8 WMMAs); wave-private LDS round-trip, DS kept in order.
    #pragma unroll
    for (int ksp = 0; ksp < 2; ++ksp) {
      Frag16 pf;
      int dd = ksp * 32 + (lhi ? 8 : 0);
      pf.h[0] = *(const bf16x8*)(ps + pbase + l16 * 72 + dd);
      pf.h[1] = *(const bf16x8*)(ps + pbase + l16 * 72 + dd + 16);
      #pragma unroll
      for (int dn = 0; dn < 4; ++dn) {
        Frag16 vf;
        int dc = dn * 16 + l16;
        int jj = ksp * 32 + (lhi ? 8 : 0);
        vf.h[0] = *(const bf16x8*)(vs + dc * 72 + jj);
        vf.h[1] = *(const bf16x8*)(vs + dc * 72 + jj + 16);
        accO[dn] = wmma_bf16(pf, vf, accO[dn]);
      }
    }
  }

  // Normalize and emit O as bf16 [bh][t][d].
  #pragma unroll
  for (int dn = 0; dn < 4; ++dn) {
    #pragma unroll
    for (int vv = 0; vv < 8; ++vv) {
      int t = qb0 + wid * 16 + vv + (lhi << 3);
      og[(size_t)t * 64 + dn * 16 + l16] = (bf16)(accO[dn][vv] / lsum[vv]);
    }
  }
}

// ---------------------------------------------------------------------------
// Kernel 3: out = attnO(reshaped) @ w_out + b_out.  M=4096, N=1024, K=1024.
// A is bf16 [BH,T,64] remapped to [m = b*T+t, k = h*64+d].
// ---------------------------------------------------------------------------
__global__ __launch_bounds__(256) void out_proj_kernel(
    const bf16* __restrict__ ao, const float* __restrict__ w,
    const float* __restrict__ bias, float* __restrict__ out) {
  __shared__ bf16 As[128 * 40];
  __shared__ bf16 Bs[128 * 40];

  const int tid  = threadIdx.x, lane = tid & 31, wid = tid >> 5;
  const int l16  = lane & 15, lhi = lane >> 4;
  const int waveM = wid & 3, waveN = wid >> 2;
  const int mBase = (blockIdx.x & 31) * 128;          // 32 M blocks
  const int nBase = (blockIdx.x >> 5) * 128;          // 8 N blocks

  v8f acc[2][4] = {};

  for (int k0 = 0; k0 < 1024; k0 += 32) {
    int h = k0 >> 6, kin = k0 & 63;                   // 32-wide K tile stays in one head
    __syncthreads();
    if (k0 + 32 < 1024) {
      __builtin_prefetch(w + (size_t)(k0 + 32 + (tid >> 5)) * 1024 + nBase + ((tid & 31) << 2), 0, 1);
    }
    // Stage A tile 128x32 from bf16 attention output (remapped indices).
    #pragma unroll
    for (int i = 0; i < 2; ++i) {
      int li = tid + i * 256;
      int m = li >> 2, dq = (li & 3) << 3;
      int mg = mBase + m, bidx = mg >> 11, t = mg & 2047;
      async_copy_b128(ao + ((size_t)(bidx * 16 + h) * 2048 + t) * 64 + kin + dq,
                      As + m * 40 + dq);
    }
    // Stage B tile (w_out f32 [K][N]) transposed to bf16 [n][k].
    #pragma unroll
    for (int i = 0; i < 4; ++i) {
      int li = tid + i * 256;
      int kk = li >> 5, nq = (li & 31) << 2;
      float4 f = *(const float4*)(w + (size_t)(k0 + kk) * 1024 + nBase + nq);
      Bs[(nq + 0) * 40 + kk] = (bf16)f.x;
      Bs[(nq + 1) * 40 + kk] = (bf16)f.y;
      Bs[(nq + 2) * 40 + kk] = (bf16)f.z;
      Bs[(nq + 3) * 40 + kk] = (bf16)f.w;
    }
    async_wait0();
    __syncthreads();

    Frag16 a[2], b[4];
    #pragma unroll
    for (int mr = 0; mr < 2; ++mr) {
      int row = waveM * 32 + mr * 16 + l16;
      int d0 = lhi ? 8 : 0;
      a[mr].h[0] = *(const bf16x8*)(As + row * 40 + d0);
      a[mr].h[1] = *(const bf16x8*)(As + row * 40 + d0 + 16);
    }
    #pragma unroll
    for (int nr = 0; nr < 4; ++nr) {
      int col = waveN * 64 + nr * 16 + l16;
      int d0 = lhi ? 8 : 0;
      b[nr].h[0] = *(const bf16x8*)(Bs + col * 40 + d0);
      b[nr].h[1] = *(const bf16x8*)(Bs + col * 40 + d0 + 16);
    }
    #pragma unroll
    for (int mr = 0; mr < 2; ++mr)
      #pragma unroll
      for (int nr = 0; nr < 4; ++nr)
        acc[mr][nr] = wmma_bf16(a[mr], b[nr], acc[mr][nr]);
  }

  #pragma unroll
  for (int nr = 0; nr < 4; ++nr) {
    int n = nBase + waveN * 64 + nr * 16 + l16;
    float bv = bias[n];
    #pragma unroll
    for (int mr = 0; mr < 2; ++mr) {
      #pragma unroll
      for (int vv = 0; vv < 8; ++vv) {
        int m = mBase + waveM * 32 + mr * 16 + vv + (lhi << 3);
        out[(size_t)m * 1024 + n] = acc[mr][nr][vv] + bv;
      }
    }
  }
}

// ---------------------------------------------------------------------------
// Launch: inputs (f32): x, w_qkv, b_qkv, w_out, b_out.  Output f32 [B,T,C].
// Workspace layout (bf16): Q(8MB) | K(8MB) | V^T(8MB) | attnO(8MB) = 32MB.
// ---------------------------------------------------------------------------
extern "C" void kernel_launch(void* const* d_in, const int* in_sizes, int n_in,
                              void* d_out, int out_size, void* d_ws, size_t ws_size,
                              hipStream_t stream) {
  (void)in_sizes; (void)n_in; (void)out_size; (void)ws_size;
  const float* x     = (const float*)d_in[0];
  const float* w_qkv = (const float*)d_in[1];
  const float* b_qkv = (const float*)d_in[2];
  const float* w_out = (const float*)d_in[3];
  const float* b_out = (const float*)d_in[4];
  float* out = (float*)d_out;

  char* ws = (char*)d_ws;
  bf16* qB  = (bf16*)(ws);
  bf16* kB  = (bf16*)(ws + ((size_t)8  << 20));
  bf16* vtB = (bf16*)(ws + ((size_t)16 << 20));
  bf16* aoB = (bf16*)(ws + ((size_t)24 << 20));

  qkv_proj_kernel<<<dim3(32 * 24), 256, 0, stream>>>(x, w_qkv, b_qkv, qB, kB, vtB);
  flash_attn_kernel<<<dim3(16, 32), 256, 0, stream>>>(qB, kB, vtB, aoB);
  out_proj_kernel<<<dim3(32 * 8), 256, 0, stream>>>(aoB, w_out, b_out, out);
}
